// DTFDynamicLayer_57174604645050
// MI455X (gfx1250) — compile-verified, matrix operand order checked by reading, not run on hardware
//
#include <hip/hip_runtime.h>
#include <math.h>

typedef __attribute__((ext_vector_type(16))) _Float16 v16h;
typedef __attribute__((ext_vector_type(8)))  _Float16 v8h;
typedef __attribute__((ext_vector_type(8)))  float    v8f;

static __device__ __forceinline__ v16h pack16(v8h lo, v8h hi) {
  return __builtin_shufflevector(lo, hi, 0,1,2,3,4,5,6,7,8,9,10,11,12,13,14,15);
}

// gfx1250 async global->LDS copy (ASYNCcnt-tracked, bypasses VGPRs).
static __device__ __forceinline__ void async_ld_b128(unsigned lds_off, const void* gptr) {
  asm volatile("global_load_async_to_lds_b128 %0, %1, off"
               :: "v"(lds_off), "v"((unsigned long long)(uintptr_t)gptr)
               : "memory");
}

// ---------------------------------------------------------------------------
// 1) Router: scores[b,t] = beta_cu*||o-p|| + beta_ce*(||p-pr|| + ce_off)
// ---------------------------------------------------------------------------
__global__ void router_kernel(const float* __restrict__ orig, const float* __restrict__ post,
                              const float* __restrict__ prior,
                              const float* beta_ce, const float* beta_cu, const float* ce_off,
                              float* __restrict__ scores, int Dd) {
  int tok = blockIdx.x;
  const float* o = orig  + (size_t)tok * Dd;
  const float* p = post  + (size_t)tok * Dd;
  const float* r = prior + (size_t)tok * Dd;
  float s_cu = 0.f, s_ce = 0.f;
  for (int i = threadIdx.x; i < Dd; i += blockDim.x) {
    float d1 = o[i] - p[i]; s_cu += d1 * d1;
    float d2 = p[i] - r[i]; s_ce += d2 * d2;
  }
  __shared__ float red0[256], red1[256];
  red0[threadIdx.x] = s_cu; red1[threadIdx.x] = s_ce;
  __syncthreads();
  for (int st = 128; st > 0; st >>= 1) {
    if (threadIdx.x < st) { red0[threadIdx.x] += red0[threadIdx.x + st];
                            red1[threadIdx.x] += red1[threadIdx.x + st]; }
    __syncthreads();
  }
  if (threadIdx.x == 0) {
    float cu = sqrtf(red0[0]), ce = sqrtf(red1[0]);
    scores[tok] = beta_cu[0] * cu + beta_ce[0] * (ce + ce_off[0]);
  }
}

// ---------------------------------------------------------------------------
// 2) Exact top-K per batch row, temporal order (matches lax.top_k tie-break)
// ---------------------------------------------------------------------------
__global__ void topk_kernel(const float* __restrict__ scores, int* __restrict__ tok_idx,
                            float* __restrict__ sel_gate, int T_, int K_) {
  __shared__ float sc[2048];
  __shared__ int   fl[2048];
  int b = blockIdx.x;
  const float* row = scores + (size_t)b * T_;
  for (int t = threadIdx.x; t < T_; t += blockDim.x) sc[t] = row[t];
  __syncthreads();
  for (int t = threadIdx.x; t < T_; t += blockDim.x) {
    float st = sc[t]; int rank = 0;
    for (int j = 0; j < T_; ++j) {
      float sj = sc[j];
      rank += (sj > st) || (sj == st && j < t);
    }
    fl[t] = (rank < K_) ? 1 : 0;
  }
  __syncthreads();
  for (int t = threadIdx.x; t < T_; t += blockDim.x) {
    if (fl[t]) {
      int pos = 0;
      for (int j = 0; j < t; ++j) pos += fl[j];
      tok_idx[b * K_ + pos]  = t;
      sel_gate[b * K_ + pos] = 1.f / (1.f + __expf(-sc[t]));
    }
  }
}

// ---------------------------------------------------------------------------
// 3) Gather selected rows + RMSNorm(ln1) -> fp16, also gather cos/sin
// ---------------------------------------------------------------------------
__global__ void gather_rms1_kernel(const float* __restrict__ hs, const float* __restrict__ cos_,
                                   const float* __restrict__ sin_, const int* __restrict__ tok_idx,
                                   const float* __restrict__ ln_w,
                                   float* __restrict__ sel_h, float* __restrict__ cosg,
                                   float* __restrict__ sing, _Float16* __restrict__ h16,
                                   int T_, int K_, int Dd, int HDd) {
  int s = blockIdx.x;
  int b = s / K_;
  int t = tok_idx[s];
  const float* src = hs + ((size_t)(b * T_ + t)) * Dd;
  float ss = 0.f;
  for (int i = threadIdx.x; i < Dd; i += blockDim.x) {
    float v = src[i]; ss += v * v;
    sel_h[(size_t)s * Dd + i] = v;
  }
  __shared__ float red[256];
  red[threadIdx.x] = ss; __syncthreads();
  for (int st = 128; st > 0; st >>= 1) {
    if (threadIdx.x < st) red[threadIdx.x] += red[threadIdx.x + st];
    __syncthreads();
  }
  float rstd = rsqrtf(red[0] / (float)Dd + 1e-6f);
  for (int i = threadIdx.x; i < Dd; i += blockDim.x)
    h16[(size_t)s * Dd + i] = (_Float16)(src[i] * rstd * ln_w[i]);
  for (int i = threadIdx.x; i < HDd; i += blockDim.x) {
    cosg[s * HDd + i] = cos_[((size_t)(b * T_ + t)) * HDd + i];
    sing[s * HDd + i] = sin_[((size_t)(b * T_ + t)) * HDd + i];
  }
}

// ---------------------------------------------------------------------------
// 4) RMSNorm(ln2) -> fp16
// ---------------------------------------------------------------------------
__global__ void rms2_kernel(const float* __restrict__ x, const float* __restrict__ ln_w,
                            _Float16* __restrict__ h16, int Dd) {
  int s = blockIdx.x;
  const float* src = x + (size_t)s * Dd;
  float ss = 0.f;
  for (int i = threadIdx.x; i < Dd; i += blockDim.x) { float v = src[i]; ss += v * v; }
  __shared__ float red[256];
  red[threadIdx.x] = ss; __syncthreads();
  for (int st = 128; st > 0; st >>= 1) {
    if (threadIdx.x < st) red[threadIdx.x] += red[threadIdx.x + st];
    __syncthreads();
  }
  float rstd = rsqrtf(red[0] / (float)Dd + 1e-6f);
  for (int i = threadIdx.x; i < Dd; i += blockDim.x)
    h16[(size_t)s * Dd + i] = (_Float16)(src[i] * rstd * ln_w[i]);
}

// ---------------------------------------------------------------------------
// 5) Weight cast+transpose: src[K,N] f32 -> dst[N,K] f16 (B operand layout)
// ---------------------------------------------------------------------------
__global__ void castT_kernel(const float* __restrict__ src, _Float16* __restrict__ dst,
                             int Kd, int Nd) {
  __shared__ float tile[32][33];
  int k0 = blockIdx.y * 32, n0 = blockIdx.x * 32;
  int tx = threadIdx.x, ty = threadIdx.y;  // 32 x 8
  for (int r = ty; r < 32; r += 8) tile[r][tx] = src[(size_t)(k0 + r) * Nd + n0 + tx];
  __syncthreads();
  for (int r = ty; r < 32; r += 8) dst[(size_t)(n0 + r) * Kd + k0 + tx] = (_Float16)tile[tx][r];
}

// ---------------------------------------------------------------------------
// 6) WMMA GEMM, LDS-staged macro-tile with double-buffered async copies.
//    Block (256 thr, 8 waves) owns a 128x256 C tile; waves in 2x4 grid each
//    compute 64x64 via v_wmma_f32_16x16x32_f16. Per K-step (32):
//      A panel 128x32 + B panel 256x32 staged via global_load_async_to_lds.
// ---------------------------------------------------------------------------
__global__ __launch_bounds__(256) void gemm16_kernel(
    const _Float16* __restrict__ A, const _Float16* __restrict__ Bt,
    const float* __restrict__ bias, const float* __restrict__ resid,
    float* __restrict__ C, int M, int N, int Kd) {
  __shared__ __align__(16) _Float16 lA[2][128 * 32];
  __shared__ __align__(16) _Float16 lB[2][256 * 32];
  int tid  = threadIdx.x;
  int lane = tid & 31;
  int wid  = tid >> 5;
  int nblk = N >> 8;                   // N/256
  int gm0  = (blockIdx.x / nblk) << 7; // *128
  int gn0  = (blockIdx.x % nblk) << 8; // *256
  int lrow  = lane & 15;
  int khalf = (lane >> 4) << 3;        // K split per 16-bit A/B operand layout
  int rbase = (lane >> 4) * 8;
  int m0w = (wid >> 2) * 64;
  int n0w = (wid & 3) * 64;

  // async-copy coordinates: chunk = 16B; A: 512 chunks, B: 1024 chunks
  int crow  = tid >> 2;        // 0..63
  int cpart = (tid & 3) * 8;   // half-offset within a 32-half row

  auto issue_tile = [&](int buf, int kk) {
    unsigned la = (unsigned)(uintptr_t)&lA[buf][0];
    unsigned lb = (unsigned)(uintptr_t)&lB[buf][0];
#pragma unroll
    for (int i = 0; i < 2; ++i) {
      int row = crow + i * 64;
      async_ld_b128(la + (unsigned)(row * 32 + cpart) * 2,
                    A + (size_t)(gm0 + row) * Kd + kk + cpart);
    }
#pragma unroll
    for (int i = 0; i < 4; ++i) {
      int row = crow + i * 64;
      async_ld_b128(lb + (unsigned)(row * 32 + cpart) * 2,
                    Bt + (size_t)(gn0 + row) * Kd + kk + cpart);
    }
  };

  v8f acc[4][4] = {};
  issue_tile(0, 0);
  int buf = 0;
  for (int kk = 0; kk < Kd; kk += 32, buf ^= 1) {
    if (kk + 32 < Kd) {
      issue_tile(buf ^ 1, kk + 32);
      // in-order completion: <=6 outstanding means current tile's 6 retired
      asm volatile("s_wait_asynccnt 0x6" ::: "memory");
    } else {
      asm volatile("s_wait_asynccnt 0x0" ::: "memory");
    }
    __syncthreads();
    const _Float16* pA = &lA[buf][0];
    const _Float16* pB = &lB[buf][0];
    v16h af[4], bf[4];
#pragma unroll
    for (int i = 0; i < 4; ++i) {
      const _Float16* ap = pA + (m0w + 16 * i + lrow) * 32 + khalf;
      af[i] = pack16(*(const v8h*)ap, *(const v8h*)(ap + 16));
    }
#pragma unroll
    for (int j = 0; j < 4; ++j) {
      const _Float16* bp = pB + (n0w + 16 * j + lrow) * 32 + khalf;
      bf[j] = pack16(*(const v8h*)bp, *(const v8h*)(bp + 16));
    }
#pragma unroll
    for (int i = 0; i < 4; ++i)
#pragma unroll
      for (int j = 0; j < 4; ++j)
        acc[i][j] = __builtin_amdgcn_wmma_f32_16x16x32_f16(
            false, af[i], false, bf[j], (short)0, acc[i][j], false, false);
    __syncthreads();  // all waves done reading buf before it is refilled
  }
#pragma unroll
  for (int i = 0; i < 4; ++i) {
#pragma unroll
    for (int j = 0; j < 4; ++j) {
      int col = gn0 + n0w + 16 * j + lrow;
      float bb = bias ? bias[col] : 0.f;
#pragma unroll
      for (int r = 0; r < 8; ++r) {
        int row = gm0 + m0w + 16 * i + rbase + r;
        float v = acc[i][j][r] + bb;
        if (resid) v += resid[(size_t)row * N + col];
        C[(size_t)row * N + col] = v;
      }
    }
  }
}

// ---------------------------------------------------------------------------
// 7) RoPE + cast: q,k f32 -> rotated fp16; v -> per-head transposed fp16
// ---------------------------------------------------------------------------
__global__ void rope_kernel(const float* __restrict__ q, const float* __restrict__ k,
                            const float* __restrict__ v, const float* __restrict__ cosg,
                            const float* __restrict__ sing,
                            _Float16* __restrict__ qr, _Float16* __restrict__ kr,
                            _Float16* __restrict__ vt, int S_, int Hh, int HDd) {
  int s = blockIdx.x;
  int half = HDd >> 1;  // 64
  for (int idx = threadIdx.x; idx < Hh * half; idx += blockDim.x) {
    int h = idx / half, d = idx % half;
    size_t base = (size_t)s * Hh * HDd + (size_t)h * HDd;
    float c0 = cosg[s * HDd + d], c1 = cosg[s * HDd + d + half];
    float s0 = sing[s * HDd + d], s1 = sing[s * HDd + d + half];
    float qa = q[base + d], qb = q[base + d + half];
    qr[base + d]        = (_Float16)(qa * c0 - qb * s0);
    qr[base + d + half] = (_Float16)(qb * c1 + qa * s1);
    float ka = k[base + d], kb = k[base + d + half];
    kr[base + d]        = (_Float16)(ka * c0 - kb * s0);
    kr[base + d + half] = (_Float16)(kb * c1 + ka * s1);
    vt[(size_t)(h * HDd + d) * S_ + s]        = (_Float16)v[base + d];
    vt[(size_t)(h * HDd + d + half) * S_ + s] = (_Float16)v[base + d + half];
  }
}

// ---------------------------------------------------------------------------
// 8) Flash attention: 1 wave = 16 queries x 128 d for one head, causal,
//    WMMA for QK^T and PV, online softmax, P re-layout through LDS.
// ---------------------------------------------------------------------------
__global__ __launch_bounds__(256) void attn_kernel(
    const _Float16* __restrict__ qr, const _Float16* __restrict__ kr,
    const _Float16* __restrict__ vt, _Float16* __restrict__ attn, int S_, int Hh) {
  const int HDd = 128;
  __shared__ __align__(16) _Float16 pbuf[8][16 * 32];
  int lane = threadIdx.x & 31;
  int wid  = threadIdx.x >> 5;
  int g  = blockIdx.x * 8 + wid;
  int h  = g >> 7;          // S_/16 == 128 q-tiles per head
  int q0 = (g & 127) * 16;
  int lrow  = lane & 15;
  int khalf = (lane >> 4) << 3;
  int rbase = (lane >> 4) * 8;
  _Float16* myp = pbuf[wid];
  int rowstride = Hh * HDd;

  v16h qf[4];
#pragma unroll
  for (int c = 0; c < 4; ++c) {
    const _Float16* qp = qr + (size_t)(q0 + lrow) * rowstride + h * HDd + c * 32 + khalf;
    qf[c] = pack16(*(const v8h*)qp, *(const v8h*)(qp + 16));
  }
  v8f o[8] = {};
  float run_m[8], run_l[8];
#pragma unroll
  for (int r = 0; r < 8; ++r) { run_m[r] = -1e30f; run_l[r] = 0.f; }
  const float scale = 0.08838834764831845f;  // 1/sqrt(128)

  for (int kb = 0; kb <= q0 + 15; kb += 32) {
    v8f s0 = {}, s1 = {};
#pragma unroll
    for (int c = 0; c < 4; ++c) {
      const _Float16* kp0 = kr + (size_t)(kb + lrow) * rowstride + h * HDd + c * 32 + khalf;
      v16h kf0 = pack16(*(const v8h*)kp0, *(const v8h*)(kp0 + 16));
      s0 = __builtin_amdgcn_wmma_f32_16x16x32_f16(false, qf[c], false, kf0, (short)0, s0, false, false);
      const _Float16* kp1 = kr + (size_t)(kb + 16 + lrow) * rowstride + h * HDd + c * 32 + khalf;
      v16h kf1 = pack16(*(const v8h*)kp1, *(const v8h*)(kp1 + 16));
      s1 = __builtin_amdgcn_wmma_f32_16x16x32_f16(false, qf[c], false, kf1, (short)0, s1, false, false);
    }
    float tmax[8];
#pragma unroll
    for (int r = 0; r < 8; ++r) {
      int qg = q0 + rbase + r;
      float a = s0[r] * scale; if (kb + lrow      > qg) a = -1e9f;
      float b = s1[r] * scale; if (kb + 16 + lrow > qg) b = -1e9f;
      s0[r] = a; s1[r] = b;
      float mx = fmaxf(a, b);
      mx = fmaxf(mx, __shfl_xor(mx, 1, 32));
      mx = fmaxf(mx, __shfl_xor(mx, 2, 32));
      mx = fmaxf(mx, __shfl_xor(mx, 4, 32));
      mx = fmaxf(mx, __shfl_xor(mx, 8, 32));
      tmax[r] = mx;
    }
    float alpha[8];
#pragma unroll
    for (int r = 0; r < 8; ++r) {
      float nm = fmaxf(run_m[r], tmax[r]);
      alpha[r] = __expf(run_m[r] - nm);
      run_m[r] = nm;
      float p0 = __expf(s0[r] - nm);
      float p1 = __expf(s1[r] - nm);
      s0[r] = p0; s1[r] = p1;
      float ps = p0 + p1;
      ps += __shfl_xor(ps, 1, 32);
      ps += __shfl_xor(ps, 2, 32);
      ps += __shfl_xor(ps, 4, 32);
      ps += __shfl_xor(ps, 8, 32);
      run_l[r] = run_l[r] * alpha[r] + ps;
    }
#pragma unroll
    for (int r = 0; r < 8; ++r) {
      int m = rbase + r;
      myp[m * 32 + lrow]      = (_Float16)s0[r];
      myp[m * 32 + 16 + lrow] = (_Float16)s1[r];
    }
    asm volatile("s_wait_dscnt 0" ::: "memory");
    const _Float16* pp = myp + lrow * 32 + khalf;
    v16h pf = pack16(*(const v8h*)pp, *(const v8h*)(pp + 16));
    asm volatile("s_wait_dscnt 0" ::: "memory");
#pragma unroll
    for (int t = 0; t < 8; ++t) {
#pragma unroll
      for (int r = 0; r < 8; ++r) o[t][r] = o[t][r] * alpha[r];
    }
#pragma unroll
    for (int t = 0; t < 8; ++t) {
      const _Float16* vp = vt + (size_t)(h * HDd + t * 16 + lrow) * S_ + kb + khalf;
      v16h vf = pack16(*(const v8h*)vp, *(const v8h*)(vp + 16));
      o[t] = __builtin_amdgcn_wmma_f32_16x16x32_f16(false, pf, false, vf, (short)0, o[t], false, false);
    }
  }
#pragma unroll
  for (int r = 0; r < 8; ++r) run_l[r] = 1.f / run_l[r];
#pragma unroll
  for (int t = 0; t < 8; ++t) {
#pragma unroll
    for (int r = 0; r < 8; ++r) {
      int m = rbase + r;
      attn[(size_t)(q0 + m) * rowstride + h * HDd + t * 16 + lrow] = (_Float16)(o[t][r] * run_l[r]);
    }
  }
}

// ---------------------------------------------------------------------------
// 9) silu(gate) * up -> fp16
// ---------------------------------------------------------------------------
__global__ void silu_mul_kernel(const float* __restrict__ g, const float* __restrict__ u,
                                _Float16* __restrict__ out, size_t n) {
  size_t i = (size_t)blockIdx.x * blockDim.x + threadIdx.x;
  size_t stride = (size_t)gridDim.x * blockDim.x;
  for (; i < n; i += stride) {
    float x = g[i];
    out[i] = (_Float16)((x / (1.f + __expf(-x))) * u[i]);
  }
}

// ---------------------------------------------------------------------------
// 10) copy + gated scatter back
// ---------------------------------------------------------------------------
__global__ void copy_kernel(const float* __restrict__ src, float* __restrict__ dst, size_t n) {
  size_t i = (size_t)blockIdx.x * blockDim.x + threadIdx.x;
  size_t stride = (size_t)gridDim.x * blockDim.x;
  for (; i < n; i += stride) dst[i] = src[i];
}

__global__ void scatter_kernel(const float* __restrict__ sel_h, const float* __restrict__ proc,
                               const float* __restrict__ gate, const int* __restrict__ tok_idx,
                               float* __restrict__ out, int T_, int K_, int Dd) {
  int s = blockIdx.x;
  int b = s / K_;
  int t = tok_idx[s];
  float gg = gate[s];
  float* dst = out + ((size_t)(b * T_ + t)) * Dd;
  const float* sh = sel_h + (size_t)s * Dd;
  const float* pr = proc  + (size_t)s * Dd;
  for (int i = threadIdx.x; i < Dd; i += blockDim.x)
    dst[i] = sh[i] + (pr[i] - sh[i]) * gg;
}

// ---------------------------------------------------------------------------
extern "C" void kernel_launch(void* const* d_in, const int* in_sizes, int n_in,
                              void* d_out, int out_size, void* d_ws, size_t ws_size,
                              hipStream_t stream) {
  (void)in_sizes; (void)n_in; (void)out_size; (void)ws_size;
  const int Bv = 2, Tv = 2048, Dv = 2048, Hv = 16, HDv = 128, Iv = 5632;
  const int Kv = Tv / 2, Sv = Bv * Kv;  // K=1024, S=2048

  const float* hs      = (const float*)d_in[0];
  const float* orig    = (const float*)d_in[1];
  const float* post    = (const float*)d_in[2];
  const float* prior   = (const float*)d_in[3];
  const float* cosi    = (const float*)d_in[4];
  const float* sini    = (const float*)d_in[5];
  /* d_in[6] position_ids is unused by the reference */
  const float* q_w     = (const float*)d_in[7];
  const float* q_b     = (const float*)d_in[8];
  const float* k_w     = (const float*)d_in[9];
  const float* k_b     = (const float*)d_in[10];
  const float* v_w     = (const float*)d_in[11];
  const float* v_b     = (const float*)d_in[12];
  const float* o_w     = (const float*)d_in[13];
  const float* ln1     = (const float*)d_in[14];
  const float* ln2     = (const float*)d_in[15];
  const float* gate_w  = (const float*)d_in[16];
  const float* up_w    = (const float*)d_in[17];
  const float* down_w  = (const float*)d_in[18];
  const float* beta_ce = (const float*)d_in[19];
  const float* beta_cu = (const float*)d_in[20];
  const float* ce_off  = (const float*)d_in[21];
  float* out = (float*)d_out;

  char* wsb = (char*)d_ws;
  size_t off = 0;
  auto alloc = [&](size_t bytes) -> void* {
    void* p = wsb + off;
    off += (bytes + 255) & ~(size_t)255;
    return p;
  };
  float*    scores   = (float*)alloc((size_t)Bv * Tv * 4);
  int*      tok_idx  = (int*)alloc((size_t)Sv * 4);
  float*    sel_gate = (float*)alloc((size_t)Sv * 4);
  float*    sel_h    = (float*)alloc((size_t)Sv * Dv * 4);
  float*    cosg     = (float*)alloc((size_t)Sv * HDv * 4);
  float*    sing     = (float*)alloc((size_t)Sv * HDv * 4);
  _Float16* h1       = (_Float16*)alloc((size_t)Sv * Dv * 2);
  _Float16* wq_t     = (_Float16*)alloc((size_t)Dv * Dv * 2);
  _Float16* wk_t     = (_Float16*)alloc((size_t)Dv * Dv * 2);
  _Float16* wv_t     = (_Float16*)alloc((size_t)Dv * Dv * 2);
  _Float16* wo_t     = (_Float16*)alloc((size_t)Dv * Dv * 2);
  _Float16* wg_t     = (_Float16*)alloc((size_t)Dv * Iv * 2);
  _Float16* wu_t     = (_Float16*)alloc((size_t)Dv * Iv * 2);
  _Float16* wd_t     = (_Float16*)alloc((size_t)Dv * Iv * 2);
  float*    qf32     = (float*)alloc((size_t)Sv * Dv * 4);
  float*    kf32     = (float*)alloc((size_t)Sv * Dv * 4);
  float*    vf32     = (float*)alloc((size_t)Sv * Dv * 4);
  _Float16* qr16     = (_Float16*)alloc((size_t)Sv * Dv * 2);
  _Float16* kr16     = (_Float16*)alloc((size_t)Sv * Dv * 2);
  _Float16* vt16     = (_Float16*)alloc((size_t)Sv * Dv * 2);
  _Float16* attn16   = (_Float16*)alloc((size_t)Sv * Dv * 2);
  float*    x1       = (float*)alloc((size_t)Sv * Dv * 4);
  _Float16* h2       = (_Float16*)alloc((size_t)Sv * Dv * 2);
  float*    gatef    = (float*)alloc((size_t)Sv * Iv * 4);
  float*    upf      = (float*)alloc((size_t)Sv * Iv * 4);
  _Float16* mlp16    = (_Float16*)alloc((size_t)Sv * Iv * 2);
  float*    proc     = (float*)alloc((size_t)Sv * Dv * 4);

  // Router + selection + gather/norm
  router_kernel<<<Bv * Tv, 256, 0, stream>>>(orig, post, prior, beta_ce, beta_cu, ce_off, scores, Dv);
  topk_kernel<<<Bv, 1024, 0, stream>>>(scores, tok_idx, sel_gate, Tv, Kv);
  gather_rms1_kernel<<<Sv, 256, 0, stream>>>(hs, cosi, sini, tok_idx, ln1,
                                             sel_h, cosg, sing, h1, Tv, Kv, Dv, HDv);
  // Weight cast + transpose to [N,K] f16
  dim3 tb(32, 8);
  castT_kernel<<<dim3(Dv / 32, Dv / 32), tb, 0, stream>>>(q_w, wq_t, Dv, Dv);
  castT_kernel<<<dim3(Dv / 32, Dv / 32), tb, 0, stream>>>(k_w, wk_t, Dv, Dv);
  castT_kernel<<<dim3(Dv / 32, Dv / 32), tb, 0, stream>>>(v_w, wv_t, Dv, Dv);
  castT_kernel<<<dim3(Dv / 32, Dv / 32), tb, 0, stream>>>(o_w, wo_t, Dv, Dv);
  castT_kernel<<<dim3(Iv / 32, Dv / 32), tb, 0, stream>>>(gate_w, wg_t, Dv, Iv);
  castT_kernel<<<dim3(Iv / 32, Dv / 32), tb, 0, stream>>>(up_w, wu_t, Dv, Iv);
  castT_kernel<<<dim3(Dv / 32, Iv / 32), tb, 0, stream>>>(down_w, wd_t, Iv, Dv);

  // QKV projections (WMMA, async-LDS macro-tiles): grid = (M/128)*(N/256)
  int gDD = (Sv / 128) * (Dv / 256);  // 128 blocks
  int gDI = (Sv / 128) * (Iv / 256);  // 352 blocks
  gemm16_kernel<<<gDD, 256, 0, stream>>>(h1, wq_t, q_b, nullptr, qf32, Sv, Dv, Dv);
  gemm16_kernel<<<gDD, 256, 0, stream>>>(h1, wk_t, k_b, nullptr, kf32, Sv, Dv, Dv);
  gemm16_kernel<<<gDD, 256, 0, stream>>>(h1, wv_t, v_b, nullptr, vf32, Sv, Dv, Dv);

  // RoPE + V transpose, then flash attention (WMMA)
  rope_kernel<<<Sv, 256, 0, stream>>>(qf32, kf32, vf32, cosg, sing, qr16, kr16, vt16, Sv, Hv, HDv);
  attn_kernel<<<(Hv * (Sv / 16)) / 8, 256, 0, stream>>>(qr16, kr16, vt16, attn16, Sv, Hv);

  // O-projection + residual
  gemm16_kernel<<<gDD, 256, 0, stream>>>(attn16, wo_t, nullptr, sel_h, x1, Sv, Dv, Dv);

  // MLP
  rms2_kernel<<<Sv, 256, 0, stream>>>(x1, ln2, h2, Dv);
  gemm16_kernel<<<gDI, 256, 0, stream>>>(h2, wg_t, nullptr, nullptr, gatef, Sv, Iv, Dv);
  gemm16_kernel<<<gDI, 256, 0, stream>>>(h2, wu_t, nullptr, nullptr, upf, Sv, Iv, Dv);
  silu_mul_kernel<<<2048, 256, 0, stream>>>(gatef, upf, mlp16, (size_t)Sv * Iv);
  gemm16_kernel<<<gDD, 256, 0, stream>>>(mlp16, wd_t, nullptr, x1, proc, Sv, Dv, Iv);

  // Assemble output
  copy_kernel<<<2048, 256, 0, stream>>>(hs, out, (size_t)Bv * Tv * Dv);
  scatter_kernel<<<Sv, 256, 0, stream>>>(sel_h, proc, sel_gate, tok_idx, out, Tv, Kv, Dv);
}
